// MockMoELayer_54778012893560
// MI455X (gfx1250) — matrix operand order repaired and checked by reading, not compiled, not run on hardware
//
#include <hip/hip_runtime.h>
#include <hip/hip_bf16.h>

#define NTOK 32768
#define DDIM 1024
#define NEXP 8

#define MT 128          // token-tile rows per block
#define NT 256          // output-dim tile per block
#define KT 32           // K chunk (bf16 wmma depth)
#define LDW 40          // LDS row stride in bf16 elems (80B, conflict-free frag loads)
#define GTHREADS 512    // 16 waves
#define NCH (DDIM / KT) // 32 K-chunks

typedef __attribute__((ext_vector_type(16))) __bf16 v16bf;
typedef __attribute__((ext_vector_type(8)))  float  v8f;
typedef unsigned short ushort_t;
typedef unsigned int   uint_t;

union FragU { uint4 q[2]; v16bf v; };

// pack two f32 -> packed bf16 pair (round-half-up + byte-perm splice)
__device__ __forceinline__ uint_t pack2(float lo, float hi) {
    union { float f; uint_t u; } a, b; a.f = lo; b.f = hi;
    return __builtin_amdgcn_perm(b.u + 0x8000u, a.u + 0x8000u, 0x07060302u);
}
__device__ __forceinline__ uint4 pack8(float4 lo, float4 hi) {
    uint4 pk;
    pk.x = pack2(lo.x, lo.y); pk.y = pack2(lo.z, lo.w);
    pk.z = pack2(hi.x, hi.y); pk.w = pack2(hi.z, hi.w);
    return pk;
}

// ---------------- Kernel 0: zero per-expert counters ----------------
__global__ void k_zero(int* __restrict__ cnt) {
    if (threadIdx.x < NEXP) cnt[threadIdx.x] = 0;
}

// ---------------- Kernel 1: gating (softmax + top-2) ----------------
// 256 threads = 8 waves; each wave handles 4 tokens; 32 tokens/block.
__global__ __launch_bounds__(256) void k_gate(
    const float* __restrict__ x, const float* __restrict__ gW,
    const float* __restrict__ gb, const float* __restrict__ eb,
    float* __restrict__ out, int* __restrict__ cnt,
    int* __restrict__ tokL, float* __restrict__ wL)
{
    __shared__ float sGW[NEXP * DDIM];              // 32 KB
    for (int i = threadIdx.x; i < NEXP * DDIM; i += 256) sGW[i] = gW[i];
    __syncthreads();

    const int lane = threadIdx.x & 31;
    const int wid  = threadIdx.x >> 5;

    for (int j = 0; j < 4; ++j) {
        const int t = blockIdx.x * 32 + wid * 4 + j;
        const float* xt = x + (size_t)t * DDIM;

        float acc[NEXP];
        #pragma unroll
        for (int e = 0; e < NEXP; ++e) acc[e] = 0.f;

        for (int i = 0; i < DDIM; i += 32) {
            const float xv = xt[i + lane];
            #pragma unroll
            for (int e = 0; e < NEXP; ++e) acc[e] += xv * sGW[e * DDIM + i + lane];
        }
        #pragma unroll
        for (int e = 0; e < NEXP; ++e) {
            #pragma unroll
            for (int off = 16; off; off >>= 1) acc[e] += __shfl_xor(acc[e], off, 32);
            acc[e] += gb[e];
        }
        // softmax over 8 logits (replicated on all lanes)
        float mx = acc[0];
        #pragma unroll
        for (int e = 1; e < NEXP; ++e) mx = fmaxf(mx, acc[e]);
        float p[NEXP]; float s = 0.f;
        #pragma unroll
        for (int e = 0; e < NEXP; ++e) { p[e] = __expf(acc[e] - mx); s += p[e]; }
        // top-2 (strict > keeps lowest index on ties, matching lax.top_k)
        int e0 = 0;
        #pragma unroll
        for (int e = 1; e < NEXP; ++e) if (p[e] > p[e0]) e0 = e;
        int e1 = (e0 == 0) ? 1 : 0;
        #pragma unroll
        for (int e = 0; e < NEXP; ++e) { if (e == e0) continue; if (p[e] > p[e1]) e1 = e; }

        const float inv = 1.f / s;
        const float w0 = p[e0] * inv, w1 = p[e1] * inv;

        if (lane == 0) {
            int q0 = atomicAdd(&cnt[e0], 1);
            tokL[e0 * NTOK + q0] = t;  wL[e0 * NTOK + q0] = w0;
            int q1 = atomicAdd(&cnt[e1], 1);
            tokL[e1 * NTOK + q1] = t;  wL[e1 * NTOK + q1] = w1;
        }
        // seed output with combined bias
        float* ot = out + (size_t)t * DDIM;
        const float* b0 = eb + (size_t)e0 * DDIM;
        const float* b1 = eb + (size_t)e1 * DDIM;
        for (int i = lane; i < DDIM; i += 32) ot[i] = w0 * b0[i] + w1 * b1[i];
    }
}

// ---------------- Kernel 2: routed expert GEMM with WMMA ----------------
// grid = (DDIM/NT, NTOK/MT, NEXP); block = 512 (16 waves)
// 128x256 block tile, 32x64 per-wave tile (8 WMMA / chunk / wave).
// Double-buffered LDS, software-pipelined global loads.
__global__ __launch_bounds__(GTHREADS) void k_gemm(
    const float* __restrict__ x, const float* __restrict__ eW,
    float* __restrict__ out, const int* __restrict__ cnt,
    const int* __restrict__ tokL, const float* __restrict__ wL)
{
    const int e  = blockIdx.z;
    const int Me = cnt[e];
    const int m0 = blockIdx.y * MT;
    if (m0 >= Me) return;
    const int n0 = blockIdx.x * NT;

    __shared__ ushort_t As[2][MT][LDW];   // 20 KB (bf16, padded rows)
    __shared__ ushort_t Bs[2][NT][LDW];   // 40 KB
    __shared__ int   sTok[MT];
    __shared__ float sWgt[MT];

    const int tid = threadIdx.x;
    if (tid < MT) {
        const int g = m0 + tid;
        if (g < Me) { sTok[tid] = tokL[e * NTOK + g]; sWgt[tid] = wL[e * NTOK + g]; }
        else        { sTok[tid] = 0;                  sWgt[tid] = 0.f; }
    }
    __syncthreads();

    const float* We = eW + (size_t)e * DDIM * DDIM;

    const int lane = tid & 31, wid = tid >> 5;
    const int mBase = (wid & 3) * 32;   // this wave's 2 M-subtiles (32 rows)
    const int nBase = (wid >> 2) * 64;  // this wave's 4 N-subtiles (64 cols)
    const int half = lane >> 4, r16 = lane & 15;

    // staging coords: A = 8 floats/thread, B = 16 floats/thread (2 rows)
    const int lr = tid >> 2;            // 0..127
    const int lc = (tid & 3) * 8;       // 0,8,16,24
    const int tokR = sTok[lr];
    const float* pax = x  + (size_t)tokR * DDIM + lc;
    const float* pb0 = We + (size_t)(n0 + lr) * DDIM + lc;
    const float* pb1 = pb0 + (size_t)128 * DDIM;

    v8f acc[2][4] = {};
    float4 ra0, ra1, rb0, rb1, rb2, rb3;   // in-flight global chunk

    auto stage = [&](int buf) {
        *(uint4*)&As[buf][lr][lc]       = pack8(ra0, ra1);
        *(uint4*)&Bs[buf][lr][lc]       = pack8(rb0, rb1);
        *(uint4*)&Bs[buf][lr + 128][lc] = pack8(rb2, rb3);
    };
    auto fetch = [&](int k) {
        ra0 = *(const float4*)(pax + k); ra1 = *(const float4*)(pax + k + 4);
        rb0 = *(const float4*)(pb0 + k); rb1 = *(const float4*)(pb0 + k + 4);
        rb2 = *(const float4*)(pb1 + k); rb3 = *(const float4*)(pb1 + k + 4);
    };

    // prologue: fetch chunk 0 and stage it into buffer 0
    fetch(0);
    stage(0);

    #pragma unroll 2
    for (int it = 0; it < NCH; ++it) {
        __syncthreads();
        // prefetch next chunk (overlaps with WMMA below)
        if (it + 1 < NCH) fetch((it + 1) * KT);
        // compute from current buffer
        {
            const int buf = it & 1;
            FragU a0f, a1f, bf[4];
            const ushort_t* p0 = &As[buf][mBase + r16][half * 8];
            a0f.q[0] = *(const uint4*)p0;  a0f.q[1] = *(const uint4*)(p0 + 16);
            const ushort_t* p1 = &As[buf][mBase + 16 + r16][half * 8];
            a1f.q[0] = *(const uint4*)p1;  a1f.q[1] = *(const uint4*)(p1 + 16);
            #pragma unroll
            for (int ni = 0; ni < 4; ++ni) {
                const ushort_t* q = &Bs[buf][nBase + ni * 16 + r16][half * 8];
                bf[ni].q[0] = *(const uint4*)q;  bf[ni].q[1] = *(const uint4*)(q + 16);
            }
            #pragma unroll
            for (int ni = 0; ni < 4; ++ni) {
                acc[0][ni] = __builtin_amdgcn_wmma_f32_16x16x32_bf16(false, a0f.v, false, bf[ni].v, (short)0, acc[0][ni], false, false);
                acc[1][ni] = __builtin_amdgcn_wmma_f32_16x16x32_bf16(false, a1f.v, false, bf[ni].v, (short)0, acc[1][ni], false, false);
            }
        }
        // stage next chunk into the other buffer
        if (it + 1 < NCH) stage((it + 1) & 1);
    }

    // epilogue: scale rows by gate weight, scatter-add into out
    const bool full = (m0 + MT) <= Me;   // uniform branch
    #pragma unroll
    for (int mi = 0; mi < 2; ++mi) {
        const int mlBase = mBase + mi * 16 + half * 8;
        int tk[8]; float wg[8];
        {
            int4   t0 = *(const int4*)&sTok[mlBase];
            int4   t1 = *(const int4*)&sTok[mlBase + 4];
            float4 w0 = *(const float4*)&sWgt[mlBase];
            float4 w1 = *(const float4*)&sWgt[mlBase + 4];
            tk[0]=t0.x; tk[1]=t0.y; tk[2]=t0.z; tk[3]=t0.w;
            tk[4]=t1.x; tk[5]=t1.y; tk[6]=t1.z; tk[7]=t1.w;
            wg[0]=w0.x; wg[1]=w0.y; wg[2]=w0.z; wg[3]=w0.w;
            wg[4]=w1.x; wg[5]=w1.y; wg[6]=w1.z; wg[7]=w1.w;
        }
        #pragma unroll
        for (int ni = 0; ni < 4; ++ni) {
            v8f a = acc[mi][ni];
            const int nG = n0 + nBase + ni * 16 + r16;
            if (full) {
                #pragma unroll
                for (int jj = 0; jj < 8; ++jj)
                    atomicAdd(&out[(size_t)tk[jj] * DDIM + nG], wg[jj] * a[jj]);
            } else {
                #pragma unroll
                for (int jj = 0; jj < 8; ++jj)
                    if (m0 + mlBase + jj < Me)
                        atomicAdd(&out[(size_t)tk[jj] * DDIM + nG], wg[jj] * a[jj]);
            }
        }
    }
}

extern "C" void kernel_launch(void* const* d_in, const int* in_sizes, int n_in,
                              void* d_out, int out_size, void* d_ws, size_t ws_size,
                              hipStream_t stream) {
    const float* x  = (const float*)d_in[0];
    const float* gW = (const float*)d_in[1];
    const float* gb = (const float*)d_in[2];
    const float* eW = (const float*)d_in[3];
    const float* eb = (const float*)d_in[4];
    float* out = (float*)d_out;

    int*   cnt  = (int*)d_ws;
    int*   tokL = (int*)((char*)d_ws + 256);
    float* wL   = (float*)((char*)d_ws + 256 + sizeof(int) * (size_t)NEXP * NTOK);

    k_zero<<<1, 32, 0, stream>>>(cnt);
    k_gate<<<NTOK / 32, 256, 0, stream>>>(x, gW, gb, eb, out, cnt, tokL, wL);
    dim3 grid(DDIM / NT, NTOK / MT, NEXP);
    k_gemm<<<grid, GTHREADS, 0, stream>>>(x, eW, out, cnt, tokL, wL);
}
